// NeRF_89790586290648
// MI455X (gfx1250) — matrix operand order repaired and checked by reading, not compiled, not run on hardware
//
#include <hip/hip_runtime.h>

typedef _Float16 h16;
typedef __attribute__((ext_vector_type(16))) _Float16 v16h;
typedef __attribute__((ext_vector_type(8)))  _Float16 v8h;
typedef __attribute__((ext_vector_type(8)))  float    v8f;
typedef __attribute__((ext_vector_type(4)))  float    v4f;

#define NFRAG 28
#define SWZ_SWAPX16 0x401f   // ds_swizzle bitmask mode: and=0x1f, or=0, xor=0x10

__device__ __forceinline__ v8f wmma_f16(v16h a, v16h b, v8f c) {
  return __builtin_amdgcn_wmma_f32_16x16x32_f16(false, a, false, b, (short)0, c, false, false);
}

__device__ __forceinline__ v16h load_w(const h16* __restrict__ wf, int frag, int lane) {
  return *(const v16h*)(wf + frag * 512 + lane * 16);
}

union H8I4 { v8h h; int i[4]; };

// Swap data between lane L and lane L^16 (cross-half-wave), no LDS memory used.
__device__ __forceinline__ v8h swap16(v8h x) {
  H8I4 u, r; u.h = x;
#pragma unroll
  for (int q = 0; q < 4; ++q) r.i[q] = __builtin_amdgcn_ds_swizzle(u.i[q], SWZ_SWAPX16);
  return r.h;
}

// f32 accumulator tile (8 rows held by this lane) -> packed f16, optional relu in f16.
__device__ __forceinline__ v8h pack_tile(v8f acc, bool relu) {
  v8h p;
#pragma unroll
  for (int r = 0; r < 8; ++r) p[r] = (h16)acc[r];
  if (relu) {
#pragma unroll
    for (int r = 0; r < 8; ++r) p[r] = (p[r] > (h16)0.0f) ? p[r] : (h16)0.0f;
  }
  return p;
}

// With row-permuted producer weights, the next-layer B fragment for K-chunk kc is
// simply concat(own tileE halves, own tileO halves) -- identical for all lanes.
__device__ __forceinline__ v16h cat2(v8h lo, v8h hi) {
  v16h b;
#pragma unroll
  for (int j = 0; j < 8; ++j) { b[j] = lo[j]; b[8 + j] = hi[j]; }
  return b;
}

// ---------------------------------------------------------------------------
// Pack weights into 28 WMMA A-fragments (16x32 MxK f16) in d_ws.
// A layout: lane<16 -> M=lane, elements j=0..15: K=ca+j (contiguous),
// ca = (lane>=16 ? 16 : 0).  K (input features) always identity order.
// For 64-wide layers feeding another WMMA, M rows are PERMUTED so that the D tile
// lands directly in next-layer B layout:
//   tile t (kc=t>>1, odd=t&1): row mr -> feature kc*32 + odd*8 + (mr<8 ? mr : 16+(mr-8))
// d1 (sigma at row 0) and c3 (color rows 0..2) keep identity rows.
// ---------------------------------------------------------------------------
__global__ __launch_bounds__(32) void pack_weights(
    const float* __restrict__ d0, const float* __restrict__ d1,
    const float* __restrict__ c0, const float* __restrict__ c1,
    const float* __restrict__ c2, const float* __restrict__ c3,
    h16* __restrict__ wf)
{
  const int frag = blockIdx.x;
  const int lane = threadIdx.x;
  const float* W; int in_dim, out_dim, mt, kc; bool perm;
  if (frag < 4)       { W = d0; in_dim = 32; out_dim = 64; mt = frag;           kc = 0;             perm = true;  }
  else if (frag < 6)  { W = d1; in_dim = 64; out_dim = 16; mt = 0;              kc = frag - 4;      perm = false; }
  else if (frag < 10) { W = c0; in_dim = 31; out_dim = 64; mt = frag - 6;       kc = 0;             perm = true;  }
  else if (frag < 18) { W = c1; in_dim = 64; out_dim = 64; mt = (frag-10) >> 1; kc = (frag-10) & 1; perm = true;  }
  else if (frag < 26) { W = c2; in_dim = 64; out_dim = 64; mt = (frag-18) >> 1; kc = (frag-18) & 1; perm = true;  }
  else                { W = c3; in_dim = 64; out_dim = 3;  mt = 0;              kc = frag - 26;     perm = false; }

  const int mr = lane & 15;
  int n;
  if (perm) {
    const int kc_out = mt >> 1, odd = mt & 1;
    n = kc_out * 32 + odd * 8 + ((mr < 8) ? mr : (16 + (mr - 8)));
  } else {
    n = mt * 16 + mr;
  }
  const int ca = (lane & 16) ? 16 : 0;
  v16h v;
#pragma unroll
  for (int j = 0; j < 16; ++j) {
    int k = kc * 32 + ca + j;
    float x = (n < out_dim && k < in_dim) ? W[n * in_dim + k] : 0.0f;
    v[j] = (h16)x;
  }
  *(v16h*)(wf + frag * 512 + lane * 16) = v;
}

// ---------------------------------------------------------------------------
// Fused MLP, all-register dataflow: one wave = 16 points, points live on lanes
// for the whole kernel.  Weight-row permutation makes every 64-wide layer
// handoff a pure register concat (no swizzle, no select); only the d1 density
// handoff needs one lane^16 swap.  28 v_wmma_f32_16x16x32_f16 per wave.
// ---------------------------------------------------------------------------
__global__ __launch_bounds__(256) void nerf_mlp(
    const float* __restrict__ hash_feat, const float* __restrict__ sh_feat,
    const h16* __restrict__ wf, float* __restrict__ out)
{
  const int  lane = threadIdx.x & 31;
  const int  wave = threadIdx.x >> 5;
  const int  p0   = blockIdx.x * 128 + wave * 16;
  const int  m    = lane & 15;                 // point index within tile
  const bool hiL  = (lane & 16) != 0;
  const int  kb   = hiL ? 16 : 0;              // B K-base for this half-wave

  // ---- B operand for d0: hash_feat row, features kb..kb+15 (contiguous) ----
  v16h b_in;
  {
    const float* hr = hash_feat + (size_t)(p0 + m) * 32 + kb;
    v4f x0 = *(const v4f*)(hr);
    v4f x1 = *(const v4f*)(hr + 4);
    v4f x2 = *(const v4f*)(hr + 8);
    v4f x3 = *(const v4f*)(hr + 12);
#pragma unroll
    for (int j = 0; j < 4; ++j) {
      b_in[j]      = (h16)x0[j];
      b_in[4 + j]  = (h16)x1[j];
      b_in[8 + j]  = (h16)x2[j];
      b_in[12 + j] = (h16)x3[j];
    }
  }

  v8h pk[4];

  // ---- d0: h = relu(hash * d0^T), 4 permuted output tiles ----
#pragma unroll
  for (int mt = 0; mt < 4; ++mt) {
    v8f acc = {};
    acc = wmma_f16(load_w(wf, mt, lane), b_in, acc);
    pk[mt] = pack_tile(acc, true);
  }

  // ---- d1: i = h * d1^T (16 feats, identity rows, no relu), K=64 chained ----
  v8f acci = {};
  {
    v16h b0 = cat2(pk[0], pk[1]);
    v16h b1 = cat2(pk[2], pk[3]);
    acci = wmma_f16(load_w(wf, 4, lane), b0, acci);
    acci = wmma_f16(load_w(wf, 5, lane), b1, acci);
  }
  const float sigma = acci[0];                 // valid on lanes 0..15 (feat 0)
  v8h pk_i = pack_tile(acci, false);
  v8h sw_i = swap16(pk_i);                     // full-EXEC swap before divergence

  // ---- B operand for c0: x = [sh_feat(16) | density(15) | 0] ----
  v16h bx;
  if (!hiL) {                                  // K = 0..15 : sh_feat row (global)
    const float* sr = sh_feat + (size_t)(p0 + m) * 16;
    v4f s0 = *(const v4f*)(sr);
    v4f s1 = *(const v4f*)(sr + 4);
    v4f s2 = *(const v4f*)(sr + 8);
    v4f s3 = *(const v4f*)(sr + 12);
#pragma unroll
    for (int j = 0; j < 4; ++j) {
      bx[j]      = (h16)s0[j];
      bx[4 + j]  = (h16)s1[j];
      bx[8 + j]  = (h16)s2[j];
      bx[12 + j] = (h16)s3[j];
    }
  } else {                                     // K = 16..31 : i[1..15], then 0
#pragma unroll
    for (int j = 0; j < 7; ++j)  bx[j] = sw_i[j + 1];   // i feats 1..7 (partner)
#pragma unroll
    for (int j = 7; j < 15; ++j) bx[j] = pk_i[j - 7];   // i feats 8..15 (own)
    bx[15] = (h16)0.0f;
  }

  // ---- c0: relu(x * c0^T), K=32, 4 permuted output tiles ----
#pragma unroll
  for (int mt = 0; mt < 4; ++mt) {
    v8f acc = {};
    acc = wmma_f16(load_w(wf, 6 + mt, lane), bx, acc);
    pk[mt] = pack_tile(acc, true);
  }

  // ---- c1, c2: relu 64->64, K=64 chained, 4 permuted output tiles each ----
#pragma unroll
  for (int layer = 0; layer < 2; ++layer) {
    const int base = 10 + layer * 8;
    v16h b0 = cat2(pk[0], pk[1]);
    v16h b1 = cat2(pk[2], pk[3]);
#pragma unroll
    for (int mt = 0; mt < 4; ++mt) {
      v8f acc = {};
      acc = wmma_f16(load_w(wf, base + mt * 2 + 0, lane), b0, acc);
      acc = wmma_f16(load_w(wf, base + mt * 2 + 1, lane), b1, acc);
      pk[mt] = pack_tile(acc, true);
    }
  }

  // ---- c3: color = x * c3^T (identity rows 0..2, padded to 16), K=64 ----
  v8f accc = {};
  {
    v16h b0 = cat2(pk[0], pk[1]);
    v16h b1 = cat2(pk[2], pk[3]);
    accc = wmma_f16(load_w(wf, 26, lane), b0, accc);
    accc = wmma_f16(load_w(wf, 27, lane), b1, accc);
  }

  // ---- store {r,g,b,sigma} : one b128 per point, lanes 0..15 ----
  if (!hiL) {
    v4f o;
    o[0] = accc[0];
    o[1] = accc[1];
    o[2] = accc[2];
    o[3] = sigma;
    *(v4f*)(out + (size_t)(p0 + m) * 4) = o;
  }
}

extern "C" void kernel_launch(void* const* d_in, const int* in_sizes, int n_in,
                              void* d_out, int out_size, void* d_ws, size_t ws_size,
                              hipStream_t stream) {
  const float* hash_feat = (const float*)d_in[0];
  const float* sh_feat   = (const float*)d_in[1];
  const float* d0 = (const float*)d_in[2];
  const float* d1 = (const float*)d_in[3];
  const float* c0 = (const float*)d_in[4];
  const float* c1 = (const float*)d_in[5];
  const float* c2 = (const float*)d_in[6];
  const float* c3 = (const float*)d_in[7];
  h16* wf = (h16*)d_ws;                        // 28 frags * 512 halves = 28 KB

  pack_weights<<<NFRAG, 32, 0, stream>>>(d0, d1, c0, c1, c2, c3, wf);

  const int N = in_sizes[0] / 32;              // 2,097,152
  const int blocks = N / 128;                  // 128 points / block (8 waves)
  nerf_mlp<<<blocks, 256, 0, stream>>>(hash_feat, sh_feat, wf, (float*)d_out);
}
